// GateAttentionLayer_85899345920378
// MI455X (gfx1250) — matrix operand order repaired and checked by reading, not compiled, not run on hardware
//
#include <hip/hip_runtime.h>
#include <hip/hip_bf16.h>
#include <math.h>

#define B_  8
#define L_  2048
#define D_  1024
#define QK_ 128
#define UV_ 2048

typedef __attribute__((ext_vector_type(16))) __bf16         v16bf;
typedef __attribute__((ext_vector_type(2)))  __bf16         v2bf;
typedef __attribute__((ext_vector_type(8)))  float          v8f;
typedef __attribute__((ext_vector_type(8)))  unsigned short ushort8;
typedef __attribute__((ext_vector_type(4)))  float          f32x4;
typedef __attribute__((ext_vector_type(2)))  float          f32x2;
typedef __attribute__((ext_vector_type(4)))  unsigned       uint4v;

__device__ __forceinline__ unsigned short f2b(float f) {
  unsigned u = __builtin_bit_cast(unsigned, f);
  u += 0x7fffu + ((u >> 16) & 1u);             // round-to-nearest-even
  return (unsigned short)(u >> 16);
}
// packed pair: lo -> [15:0], hi -> [31:16]
__device__ __forceinline__ unsigned f2b2(float lo, float hi) {
#if __has_builtin(__builtin_amdgcn_cvt_pk_bf16_f32)
  v2bf p = __builtin_amdgcn_cvt_pk_bf16_f32(lo, hi);
  return __builtin_bit_cast(unsigned, p);
#else
  return ((unsigned)f2b(hi) << 16) | (unsigned)f2b(lo);
#endif
}
__device__ __forceinline__ float b2f(unsigned short s) {
  unsigned u = ((unsigned)s) << 16;
  return __builtin_bit_cast(float, u);
}
__device__ __forceinline__ float gelu_tanh(float x) {
  float x3 = x * x * x;
  return 0.5f * x * (1.0f + tanhf(0.7978845608028654f * (x + 0.044715f * x3)));
}

// CDNA5 async DMA: copy 16B global -> LDS without touching VGPRs (ASYNCcnt-tracked).
__device__ __forceinline__ void async_ld_b128(void* lds_dst, const void* gsrc) {
  unsigned lds_off = (unsigned)(unsigned long long)lds_dst;  // low 32 bits = LDS address
  asm volatile("global_load_async_to_lds_b128 %0, %1, off"
               :: "v"(lds_off), "v"(gsrc) : "memory");
}
// Wait until at most N async ops remain in flight (in-order retirement).
template <int N>
__device__ __forceinline__ void wait_async() {
  if constexpr (N == 0)      asm volatile("s_wait_asynccnt 0x0" ::: "memory");
  else if constexpr (N == 2) asm volatile("s_wait_asynccnt 0x2" ::: "memory");
  else                       asm volatile("s_wait_asynccnt 0x4" ::: "memory");
}

enum { EPI_GELU_F32 = 0, EPI_GELU_BF16 = 1, EPI_SCALE_F32 = 2, EPI_MUL_BF16 = 3, EPI_BIAS_F32 = 4 };

// One-time: f32 [R][C] row-major -> bf16 [C][R] (LDS-tiled transpose, coalesced both sides)
__global__ void __launch_bounds__(256)
cvt_tr_bf16_k(const float* __restrict__ s, unsigned short* __restrict__ d, int R, int C) {
  __shared__ unsigned short t[32][33];
  int tx = threadIdx.x & 31;
  int ty = threadIdx.x >> 5;   // 0..7
  int c0 = blockIdx.x * 32;
  int r0 = blockIdx.y * 32;
#pragma unroll
  for (int i = 0; i < 4; ++i) {
    int r = ty * 4 + i;
    t[r][tx] = f2b(s[(size_t)(r0 + r) * C + (c0 + tx)]);
  }
  __syncthreads();
#pragma unroll
  for (int i = 0; i < 4; ++i) {
    int c = ty * 4 + i;
    d[(size_t)(c0 + c) * R + (r0 + tx)] = t[tx][c];
  }
}

// bf16 [R][C] -> bf16 [C][R], batched over z (LDS-tiled)
__global__ void __launch_bounds__(256)
tr_u16_k(const unsigned short* __restrict__ s, unsigned short* __restrict__ d, int R, int C) {
  __shared__ unsigned short t[32][33];
  const size_t zo = (size_t)blockIdx.z * R * C;
  s += zo; d += zo;
  int tx = threadIdx.x & 31;
  int ty = threadIdx.x >> 5;
  int c0 = blockIdx.x * 32;
  int r0 = blockIdx.y * 32;
#pragma unroll
  for (int i = 0; i < 4; ++i) {
    int r = ty * 4 + i;
    t[r][tx] = s[(size_t)(r0 + r) * C + (c0 + tx)];
  }
  __syncthreads();
#pragma unroll
  for (int i = 0; i < 4; ++i) {
    int c = ty * 4 + i;
    d[(size_t)(c0 + c) * R + (r0 + tx)] = t[tx][c];
  }
}

// Stage one 128x32 A tile + 32x128 B tile (fragment-swizzled) into the given LDS buffers.
template <bool A_BF16, int K, int LDA, int LDB>
__device__ __forceinline__ void stage_tiles(
    unsigned short* __restrict__ Ab, unsigned short* __restrict__ Bb,
    const float* __restrict__ Af, const unsigned short* __restrict__ Au,
    const unsigned short* __restrict__ Bu,
    int m0, int n0, int k0, int tid) {
  // A tile: 2 chunks of 8 per thread
#pragma unroll
  for (int j = 0; j < 2; ++j) {
    int idx = tid + j * 256;
    int r = idx >> 2;                // row 0..127
    int q = idx & 3;                 // 8-wide k chunk
    unsigned short* dst =
        &Ab[((((r >> 4) * 32) + (q & 1) * 16 + (r & 15)) << 4) + (q >> 1) * 8];
    if (A_BF16) {
      async_ld_b128(dst, &Au[(size_t)(m0 + r) * LDA + (k0 + q * 8)]);
    } else {
      const float* s = &Af[(size_t)(m0 + r) * LDA + (k0 + q * 8)];
      f32x4 lo = *(const f32x4*)s;
      f32x4 hi = *(const f32x4*)(s + 4);
      uint4v p;
      p[0] = f2b2(lo[0], lo[1]);
      p[1] = f2b2(lo[2], lo[3]);
      p[2] = f2b2(hi[0], hi[1]);
      p[3] = f2b2(hi[2], hi[3]);
      *(uint4v*)dst = p;
      if (k0 + 32 < K)
        __builtin_prefetch(&Af[(size_t)(m0 + r) * LDA + (k0 + 32 + q * 8)], 0, 1);
    }
  }
  // B tile (NxK, k contiguous): 2 async b128 per thread
#pragma unroll
  for (int j = 0; j < 2; ++j) {
    int id = tid + j * 256;          // chunk id 0..511
    int n  = id & 127;
    int kq = id >> 7;                // 0..3
    unsigned short* dst =
        &Bb[(((n >> 4) * 32 + (kq & 1) * 16 + (n & 15)) << 4) + (kq >> 1) * 8];
    async_ld_b128(dst, &Bu[(size_t)(n0 + n) * LDB + (k0 + kq * 8)]);
  }
}

// Block-tiled bf16 WMMA GEMM: C[M,N] = epilogue(A[M,K] * B^T) with B stored NxK (k contiguous).
// Double-buffered LDS + CDNA5 async DMA staging: tile i+1 is in flight (ASYNCcnt) while
// tile i is consumed by WMMA; s_wait_asynccnt <= N (in-order retirement) proves tile i landed.
// Block = 256 threads = 8 waves. Block tile 128x128, wave tile 32x64 (8 accums), K step 32.
template <bool A_BF16, int EPI, int K, int LDA, int LDB, int LDC>
__global__ void __launch_bounds__(256)
gemm_wmma_k(const void* __restrict__ Ap, const unsigned short* __restrict__ Bp,
            void* __restrict__ Cp, const float* __restrict__ bias,
            const unsigned short* __restrict__ gate,
            long long sA, long long sB, long long sC, long long sG,
            float scale) {
  __shared__ __align__(32) unsigned short A_lds[2][8 * 32 * 16];  // ping-pong, 16 KiB
  __shared__ __align__(32) unsigned short B_lds[2][8 * 32 * 16];  // ping-pong, 16 KiB

  const int z = blockIdx.z;
  const float*          Af = (const float*)Ap          + (size_t)z * sA;
  const unsigned short* Au = (const unsigned short*)Ap + (size_t)z * sA;
  const unsigned short* Bu = Bp + (size_t)z * sB;
  float*          Cf = (float*)Cp          + (size_t)z * sC;
  unsigned short* Cu = (unsigned short*)Cp + (size_t)z * sC;
  const unsigned short* Gp = gate ? gate + (size_t)z * sG : (const unsigned short*)0;

  const int m0   = blockIdx.y * 128;
  const int n0   = blockIdx.x * 128;
  const int tid  = threadIdx.x;
  const int lane = tid & 31;
  const int wave = tid >> 5;
  const int wm   = wave & 3;   // 32-row strip (2 row-tiles)
  const int wn   = wave >> 2;  // 64-col strip (4 col-tiles)
  const int ln   = lane & 15;
  const int lhi  = lane >> 4;

  constexpr int NPEND = A_BF16 ? 4 : 2;  // per-wave async ops per staged tile

  v8f acc[2][4] = {};

  // prologue: stage tile 0 into buffer 0
  stage_tiles<A_BF16, K, LDA, LDB>(A_lds[0], B_lds[0], Af, Au, Bu, m0, n0, 0, tid);

  for (int k0 = 0; k0 < K; k0 += 32) {
    const int buf = (k0 >> 5) & 1;
    if (k0 + 32 < K) {
      __syncthreads();  // readers of buf^1 (iteration i-1) are done
      stage_tiles<A_BF16, K, LDA, LDB>(A_lds[buf ^ 1], B_lds[buf ^ 1],
                                       Af, Au, Bu, m0, n0, k0 + 32, tid);
      wait_async<NPEND>();  // tile i landed; tile i+1 still in flight
    } else {
      wait_async<0>();      // drain: final tile landed
    }
    __syncthreads();        // everyone's tile i is visible

    // ---- fragments: 32 contiguous bytes per lane -> 2x ds_load_b128 each ----
    const unsigned short* Ab = A_lds[buf];
    const unsigned short* Bb = B_lds[buf];
    v16bf a0 = *(const v16bf*)&Ab[((2 * wm + 0) * 32 + lane) << 4];
    v16bf a1 = *(const v16bf*)&Ab[((2 * wm + 1) * 32 + lane) << 4];
#pragma unroll
    for (int tj = 0; tj < 4; ++tj) {
      v16bf bf = *(const v16bf*)&Bb[((4 * wn + tj) * 32 + lane) << 4];
      acc[0][tj] = __builtin_amdgcn_wmma_f32_16x16x32_bf16(false, a0, false, bf, (short)0, acc[0][tj], false, false);
      acc[1][tj] = __builtin_amdgcn_wmma_f32_16x16x32_bf16(false, a1, false, bf, (short)0, acc[1][tj], false, false);
    }
  }

  // ---- epilogue: C/D layout M = lhi*8 + r, N = ln ----
#pragma unroll
  for (int ti = 0; ti < 2; ++ti) {
#pragma unroll
    for (int tj = 0; tj < 4; ++tj) {
#pragma unroll
      for (int r = 0; r < 8; ++r) {
        int mg = m0 + (2 * wm + ti) * 16 + lhi * 8 + r;
        int ng = n0 + (4 * wn + tj) * 16 + ln;
        float x = acc[ti][tj][r];
        if (EPI == EPI_GELU_F32 || EPI == EPI_GELU_BF16) {
          x = gelu_tanh(x + bias[ng]);
        } else if (EPI == EPI_SCALE_F32) {
          x *= scale;
        } else if (EPI == EPI_MUL_BF16) {
          x *= b2f(Gp[(size_t)mg * LDC + ng]);
        } else {  // EPI_BIAS_F32
          x += bias[ng];
        }
        if (EPI == EPI_GELU_BF16 || EPI == EPI_MUL_BF16)
          Cu[(size_t)mg * LDC + ng] = f2b(x);
        else
          Cf[(size_t)mg * LDC + ng] = x;
      }
    }
  }
}

// q/k scale-shift + RoPE, emitting bf16 operands (packed pair stores).
__global__ void __launch_bounds__(256)
rope_qk_k(const float* __restrict__ qk,
          const float* __restrict__ gq, const float* __restrict__ betq,
          const float* __restrict__ gk, const float* __restrict__ betk,
          unsigned short* __restrict__ qo, unsigned short* __restrict__ ko) {
  int idx  = blockIdx.x * 256 + threadIdx.x;  // (B*L)*64 pairs
  int pair = idx & 63;
  int row  = idx >> 6;
  if (row >= B_ * L_) return;
  int l = row & (L_ - 1);  // position within sequence
  const float* src = qk + (size_t)row * QK_;
  int f0 = 2 * pair, f1 = f0 + 1;
  float inv = powf(10000.0f, -((float)f0) / (float)QK_);
  float fr = (float)l * inv;
  float s, c;
  sincosf(fr, &s, &c);
  float q1 = src[f0] * gq[f0] + betq[f0];
  float q2 = src[f1] * gq[f1] + betq[f1];
  float k1 = src[f0] * gk[f0] + betk[f0];
  float k2 = src[f1] * gk[f1] + betk[f1];
  size_t o = (size_t)row * QK_;
  *(unsigned*)&qo[o + f0] = f2b2(q1 * c - q2 * s, q1 * s + q2 * c);
  *(unsigned*)&ko[o + f0] = f2b2(k1 * c - k2 * s, k1 * s + k2 * c);
}

// Row softmax over L: normalizes attn f32 in place (it's a returned output)
// and writes a bf16 copy for the attn*v GEMM.
__global__ void __launch_bounds__(256)
softmax_k(float* __restrict__ attn, unsigned short* __restrict__ attn_b, int Lr) {
  __shared__ float red[256];
  int row = blockIdx.x;
  float* p = attn + (size_t)row * Lr;
  unsigned short* pb = attn_b + (size_t)row * Lr;
  int tid = threadIdx.x;

  float mx = -3.0e38f;
  for (int j = tid; j < Lr; j += 256) mx = fmaxf(mx, p[j]);
  red[tid] = mx;
  __syncthreads();
  for (int s2 = 128; s2 > 0; s2 >>= 1) {
    if (tid < s2) red[tid] = fmaxf(red[tid], red[tid + s2]);
    __syncthreads();
  }
  mx = red[0];
  __syncthreads();

  float sum = 0.f;
  for (int j = tid; j < Lr; j += 256) {
    float e = expf(p[j] - mx);
    p[j] = e;
    sum += e;
  }
  red[tid] = sum;
  __syncthreads();
  for (int s2 = 128; s2 > 0; s2 >>= 1) {
    if (tid < s2) red[tid] += red[tid + s2];
    __syncthreads();
  }
  float inv = 1.0f / red[0];
  __syncthreads();

  for (int j = tid * 2; j < Lr; j += 512) {
    float a0 = p[j] * inv;
    float a1 = p[j + 1] * inv;
    f32x2 w; w[0] = a0; w[1] = a1;
    *(f32x2*)&p[j] = w;
    *(unsigned*)&pb[j] = f2b2(a0, a1);
  }
}

extern "C" void kernel_launch(void* const* d_in, const int* in_sizes, int n_in,
                              void* d_out, int out_size, void* d_ws, size_t ws_size,
                              hipStream_t stream) {
  const float* u       = (const float*)d_in[0];
  const float* queries = (const float*)d_in[1];
  // d_in[2] = keys: unused by the reference math (k derives from queries' qk)
  const float* values  = (const float*)d_in[3];
  const float* Wqk  = (const float*)d_in[4];
  const float* bqk  = (const float*)d_in[5];
  const float* gq   = (const float*)d_in[6];
  const float* betq = (const float*)d_in[7];
  const float* gk   = (const float*)d_in[8];
  const float* betk = (const float*)d_in[9];
  const float* Wv   = (const float*)d_in[10];
  const float* bv   = (const float*)d_in[11];
  const float* Wu   = (const float*)d_in[12];
  const float* bu   = (const float*)d_in[13];
  const float* Wo   = (const float*)d_in[14];
  const float* bo   = (const float*)d_in[15];
  (void)in_sizes; (void)n_in; (void)out_size; (void)ws_size;

  const int Mtot = B_ * L_;                       // 16384
  float* o_out = (float*)d_out;                   // [16384,1024] f32
  float* attn  = o_out + (size_t)Mtot * D_;       // [8,2048,2048] f32 (output slot doubles as score scratch)

  char* w = (char*)d_ws;
  float*          qkbuf = (float*)w;           w += (size_t)Mtot * QK_ * 4;
  unsigned short* qb    = (unsigned short*)w;  w += (size_t)Mtot * QK_ * 2;
  unsigned short* kb    = (unsigned short*)w;  w += (size_t)Mtot * QK_ * 2;
  unsigned short* vb    = (unsigned short*)w;  w += (size_t)Mtot * UV_ * 2;  // v row-major
  unsigned short* vtb   = (unsigned short*)w;  w += (size_t)Mtot * UV_ * 2;  // v^T per batch
  unsigned short* ugb   = (unsigned short*)w;  w += (size_t)Mtot * UV_ * 2;
  unsigned short* ab    = (unsigned short*)w;  w += (size_t)Mtot * L_ * 2;
  unsigned short* gb    = (unsigned short*)w;  w += (size_t)Mtot * UV_ * 2;
  unsigned short* wqkT  = (unsigned short*)w;  w += (size_t)QK_ * D_ * 2;   // [128][1024]
  unsigned short* wvT   = (unsigned short*)w;  w += (size_t)UV_ * D_ * 2;   // [2048][1024]
  unsigned short* wuT   = (unsigned short*)w;  w += (size_t)UV_ * D_ * 2;   // [2048][1024]
  unsigned short* woT   = (unsigned short*)w;  w += (size_t)D_ * UV_ * 2;   // [1024][2048]

  dim3 blk(256, 1, 1);

  // 0) one-time weight convert+transpose to bf16 NxK (k contiguous for B staging)
  cvt_tr_bf16_k<<<dim3(QK_ / 32, D_ / 32, 1), blk, 0, stream>>>(Wqk, wqkT, D_, QK_);
  cvt_tr_bf16_k<<<dim3(UV_ / 32, D_ / 32, 1), blk, 0, stream>>>(Wv, wvT, D_, UV_);
  cvt_tr_bf16_k<<<dim3(UV_ / 32, D_ / 32, 1), blk, 0, stream>>>(Wu, wuT, D_, UV_);
  cvt_tr_bf16_k<<<dim3(D_ / 32, UV_ / 32, 1), blk, 0, stream>>>(Wo, woT, UV_, D_);

  // 1) qk = gelu(queries @ Wqk + bqk)  [16384 x 128], K=1024
  gemm_wmma_k<false, EPI_GELU_F32, D_, D_, D_, QK_>
      <<<dim3(QK_ / 128, Mtot / 128, 1), blk, 0, stream>>>(
      queries, wqkT, qkbuf, bqk, 0, 0, 0, 0, 0, 1.0f);

  // 2) RoPE(scale-shift(qk)) -> q,k bf16
  rope_qk_k<<<dim3((Mtot * 64) / 256, 1, 1), blk, 0, stream>>>(
      qkbuf, gq, betq, gk, betk, qb, kb);

  // 3) v = gelu(values @ Wv + bv) -> bf16  [16384 x 2048], K=1024
  gemm_wmma_k<false, EPI_GELU_BF16, D_, D_, D_, UV_>
      <<<dim3(UV_ / 128, Mtot / 128, 1), blk, 0, stream>>>(
      values, wvT, vb, bv, 0, 0, 0, 0, 0, 1.0f);

  // 3b) v^T per batch (k-contiguous B operand for attn@v)
  tr_u16_k<<<dim3(UV_ / 32, L_ / 32, B_), blk, 0, stream>>>(vb, vtb, L_, UV_);

  // 4) ug = gelu(u @ Wu + bu) -> bf16
  gemm_wmma_k<false, EPI_GELU_BF16, D_, D_, D_, UV_>
      <<<dim3(UV_ / 128, Mtot / 128, 1), blk, 0, stream>>>(
      u, wuT, ugb, bu, 0, 0, 0, 0, 0, 1.0f);

  // 5) scores = q @ k^T / sqrt(128) -> attn slot (f32), batched over B
  gemm_wmma_k<true, EPI_SCALE_F32, QK_, QK_, QK_, L_>
      <<<dim3(L_ / 128, L_ / 128, B_), blk, 0, stream>>>(
      qb, kb, attn, 0, 0,
      (long long)L_ * QK_, (long long)L_ * QK_, (long long)L_ * L_, 0,
      0.08838834764831845f);

  // 6) softmax rows (in place) + bf16 copy
  softmax_k<<<dim3(Mtot, 1, 1), blk, 0, stream>>>(attn, ab, L_);

  // 7) gated = ug * (attn @ v) -> bf16, batched over B, K=2048 (B = v^T)
  gemm_wmma_k<true, EPI_MUL_BF16, L_, L_, L_, UV_>
      <<<dim3(UV_ / 128, L_ / 128, B_), blk, 0, stream>>>(
      ab, vtb, gb, 0, ugb,
      (long long)L_ * L_, (long long)L_ * UV_, (long long)L_ * UV_, (long long)L_ * UV_,
      1.0f);

  // 8) o = gated @ Wo + bo -> f32 output  [16384 x 1024], K=2048 (B = Wo^T)
  gemm_wmma_k<true, EPI_BIAS_F32, UV_, UV_, UV_, D_>
      <<<dim3(D_ / 128, Mtot / 128, 1), blk, 0, stream>>>(
      gb, woT, o_out, bo, 0, 0, 0, 0, 0, 1.0f);
}
